// SWINAttentionLayer_9964324127494
// MI455X (gfx1250) — matrix operand order repaired and checked by reading, not compiled, not run on hardware
//
#include <hip/hip_runtime.h>

typedef __attribute__((ext_vector_type(16))) _Float16 v16h;
typedef __attribute__((ext_vector_type(8)))  _Float16 v8h;
typedef __attribute__((ext_vector_type(8)))  float    v8f;

// ---------------- problem constants ----------------
constexpr int kM    = 32;    // images
constexpr int kHW   = 64;    // H == W
constexpr int kD    = 256;   // model dim
constexpr int kNH   = 8;     // heads
constexpr int kHD   = 32;    // head dim
constexpr int kWS   = 64;    // window sequence (8x8)
constexpr int kQKV  = 768;   // 3*D

// ---------------- LDS layout (bytes) ----------------
constexpr int LDA   = 264;   // halfs; row stride for 64x256 f16 tiles (bank-conflict-free)
constexpr int LDVT  = 72;    // halfs; row stride for Vt (256 x 64)
constexpr int LDS_S = 65;    // floats; row stride for scores
constexpr int LDP   = 72;    // halfs; row stride for probability tiles

constexpr int OFF_AIN = 0;
constexpr int OFF_Q   = OFF_AIN + kWS * LDA * 2;          //  33792
constexpr int OFF_K   = OFF_Q   + kWS * LDA * 2;          //  67584
constexpr int OFF_VT  = OFF_K   + kWS * LDA * 2;          // 101376
constexpr int OFF_S   = OFF_VT  + kD  * LDVT * 2;         // 138240
constexpr int OFF_RS  = OFF_S   + kNH * kWS * LDS_S * 4;  // 271360
constexpr int SMEM_BYTES = OFF_RS + kNH * kWS * 4;        // 273408
constexpr int OFF_AOUT = OFF_S;   // scores region reused for attention output (after barrier)
// stage-0 fp32 window staging (64 KB) also overlays OFF_S (dead until stage 2)
// per-wave f16 prob tiles overlay Ain/Q (dead after scores): byte off = wv*kWS*LDP*2

// ---------------- gfx1250 async global->LDS helpers ----------------
// GLOBAL_LOAD_ASYNC_TO_LDS_B128: per-lane 16B gather, LDS address from VDST, global address from VADDR,
// tracked by ASYNCcnt (ISA 10 / async_tensor 4.4). Inline asm = portable across toolchains.
__device__ __forceinline__ void async_g2l_b128(unsigned lds_byte_addr, const void* gptr) {
  asm volatile("global_load_async_to_lds_b128 %0, %1, off"
               :: "v"(lds_byte_addr), "v"((unsigned long long)(size_t)gptr)
               : "memory");
}
__device__ __forceinline__ void wait_asynccnt0() {
  asm volatile("s_wait_asynccnt 0x0" ::: "memory");
}

// ---------------- WMMA fragment helpers ----------------
// A-matrix 16x32 f16 (ISA 7.12.2): lane L<16 -> row M=L, K = k0+hi*8+{0..7} and +16
__device__ __forceinline__ v16h frag_a(const _Float16* base, int ld, int row0, int k0, int lane) {
  const int hl = lane & 15, hi = lane >> 4;
  const _Float16* p = base + (row0 + hl) * ld + k0 + hi * 8;
  v8h lo  = *(const v8h*)p;
  v8h hi8 = *(const v8h*)(p + 16);
  return __builtin_shufflevector(lo, hi8, 0,1,2,3,4,5,6,7,8,9,10,11,12,13,14,15);
}
// B-matrix 32x16 f16: lane = column N; K = k0+hi*8+{0..7} and +16 (matches frag_a)
// `base` is column-major storage: contiguous K per column (stride ld).
__device__ __forceinline__ v16h frag_b(const _Float16* base, int ld, int col0, int k0, int lane) {
  const int hl = lane & 15, hi = lane >> 4;
  const _Float16* p = base + (col0 + hl) * ld + k0 + hi * 8;
  v8h lo  = *(const v8h*)p;
  v8h hi8 = *(const v8h*)(p + 16);
  return __builtin_shufflevector(lo, hi8, 0,1,2,3,4,5,6,7,8,9,10,11,12,13,14,15);
}
__device__ __forceinline__ v8f wmma32(v16h a, v16h b, v8f c) {
  return __builtin_amdgcn_wmma_f32_16x16x32_f16(false, a, false, b, (short)0, c, false, false);
}
// swin shift-mask region id on the un-rolled coordinate grid
__device__ __forceinline__ int region9(int gh, int gw) {
  int r = (gh < 56) ? 0 : ((gh < 60) ? 1 : 2);
  int c = (gw < 56) ? 0 : ((gw < 60) ? 1 : 2);
  return r * 3 + c;
}

// ---------------- weight transpose + f16 convert ----------------
__global__ void convert_weights(const float* __restrict__ wqkv, const float* __restrict__ wo,
                                _Float16* __restrict__ wqkv_t, _Float16* __restrict__ wo_t) {
  const int idx = blockIdx.x * 256 + threadIdx.x;
  if (idx < kQKV * kD) {
    const int n = idx >> 8, k = idx & 255;
    wqkv_t[idx] = (_Float16)wqkv[k * kQKV + n];
  }
  if (idx < kD * kD) {
    const int n = idx >> 8, k = idx & 255;
    wo_t[idx] = (_Float16)wo[k * kD + n];
  }
}

// ---------------- fused SWIN attention pass ----------------
// One workgroup (8 wave32) per window. SHIFTED=1 applies the +/-4 roll and region mask.
template <int SHIFTED>
__global__ __launch_bounds__(256, 1)
void swin_pass(const float* __restrict__ xin,
               const _Float16* __restrict__ wqkv_t,  // [768][256] f16 (N-major)
               const float* __restrict__ bqkv,
               const _Float16* __restrict__ wo_t,    // [256][256] f16 (N-major)
               const float* __restrict__ bo,
               float* __restrict__ yout) {
  extern __shared__ __align__(16) char smem[];
  _Float16* Ain = (_Float16*)(smem + OFF_AIN);
  _Float16* Qs  = (_Float16*)(smem + OFF_Q);
  _Float16* Ks  = (_Float16*)(smem + OFF_K);
  _Float16* Vt  = (_Float16*)(smem + OFF_VT);

  const int t    = threadIdx.x;
  const int lane = t & 31;
  const int wv   = t >> 5;       // wave id == head id
  const int hl   = lane & 15;
  const int hi   = lane >> 4;

  const int wb = blockIdx.x;
  const int mi = wb >> 6;        // image
  const int wi = wb & 63;
  const int wh = wi >> 3;        // window row
  const int ww = wi & 7;         // window col

  const v8f vzero = {0.f,0.f,0.f,0.f,0.f,0.f,0.f,0.f};

  // L2 warm hint for the resident weights
  __builtin_prefetch(wqkv_t + (size_t)t * 768, 0, 1);
  __builtin_prefetch(wo_t   + (size_t)t * 256, 0, 1);

  // ---- stage 0a: async-gather the 64x256 fp32 window into LDS staging ----
  const int s0 = t >> 2;                 // window row 0..63
  const int q0 = t & 3;                  // quarter of the 256-wide row
  {
    int gh = wh * 8 + (s0 >> 3);
    int gw = ww * 8 + (s0 & 7);
    if (SHIFTED) { gh = (gh + 4) & 63; gw = (gw + 4) & 63; }
    const float4* src = (const float4*)(xin + (((size_t)mi * kHW + gh) * kHW + gw) * kD) + q0 * 16;
    float* stage = (float*)(smem + OFF_S) + s0 * kD + q0 * 64;   // fp32 staging in S region
    const unsigned lds0 = (unsigned)(size_t)stage;               // generic addr low 32 = LDS offset
#pragma unroll
    for (int v = 0; v < 16; ++v)
      async_g2l_b128(lds0 + v * 16, src + v);                    // ASYNCcnt-tracked
    wait_asynccnt0();                                            // own wave's asyncs done
  }
  __syncthreads();

  // ---- stage 0b: fp32 -> f16 convert into the WMMA A-tile ----
  {
    const float4* stage4 = (const float4*)((float*)(smem + OFF_S) + s0 * kD + q0 * 64);
    _Float16* dst = Ain + s0 * LDA + q0 * 64;
#pragma unroll
    for (int v = 0; v < 16; ++v) {
      float4 f = stage4[v];
      dst[4*v+0] = (_Float16)f.x; dst[4*v+1] = (_Float16)f.y;
      dst[4*v+2] = (_Float16)f.z; dst[4*v+3] = (_Float16)f.w;
    }
  }
  __syncthreads();

  // ---- stage 1: QKV projection [64x256] x [256x768] (wave owns 96 columns) ----
#pragma unroll 1
  for (int nt = 0; nt < 6; ++nt) {
    const int n0 = wv * 96 + nt * 16;
    v8f acc[4] = {vzero, vzero, vzero, vzero};
#pragma unroll
    for (int k0 = 0; k0 < kD; k0 += 32) {
      v16h b = frag_b(wqkv_t, kD, n0, k0, lane);
#pragma unroll
      for (int mt = 0; mt < 4; ++mt) {
        v16h a = frag_a(Ain, LDA, mt * 16, k0, lane);
        acc[mt] = wmma32(a, b, acc[mt]);
      }
    }
    const int n = n0 + hl;                // global qkv column (uniform split per 16-tile)
    const float bias = bqkv[n];
#pragma unroll
    for (int mt = 0; mt < 4; ++mt)
#pragma unroll
      for (int e = 0; e < 8; ++e) {
        const int m = mt * 16 + hi * 8 + e;
        const float v = acc[mt][e] + bias;
        if (n < 256)      Qs[m * LDA + n] = (_Float16)v;
        else if (n < 512) Ks[m * LDA + (n - 256)] = (_Float16)v;
        else              Vt[(n - 512) * LDVT + m] = (_Float16)v;   // V stored transposed
      }
  }
  __syncthreads();

  // ---- stage 2: scores = Q K^T * scale (+ mask), wave == head ----
  const int h = wv;
  float* Sw = (float*)(smem + OFF_S) + wv * (kWS * LDS_S);
  {
    const float scale = 0.17677669529663687f;   // 1/sqrt(32)
#pragma unroll 1
    for (int nt = 0; nt < 4; ++nt) {
      v16h b = frag_b(Ks, LDA, nt * 16, h * kHD, lane);   // column n = seq index
      v8f acc[4] = {vzero, vzero, vzero, vzero};
#pragma unroll
      for (int mt = 0; mt < 4; ++mt) {
        v16h a = frag_a(Qs, LDA, mt * 16, h * kHD, lane);
        acc[mt] = wmma32(a, b, acc[mt]);
      }
      const int n = nt * 16 + hl;
      int rk = 0;
      if (SHIFTED) rk = region9(wh * 8 + (n >> 3), ww * 8 + (n & 7));
#pragma unroll
      for (int mt = 0; mt < 4; ++mt)
#pragma unroll
        for (int e = 0; e < 8; ++e) {
          const int m = mt * 16 + hi * 8 + e;
          float v = acc[mt][e] * scale;
          if (SHIFTED) {
            const int rq = region9(wh * 8 + (m >> 3), ww * 8 + (m & 7));
            if (rq != rk) v += -1e9f;
          }
          Sw[m * LDS_S + n] = v;
        }
    }
  }
  __syncthreads();   // Q/K now dead -> prob tiles may overlay them

  // ---- stage 3: row softmax (each lane owns 2 rows), f16 probs to overlay LDS ----
  _Float16* Pw = (_Float16*)smem + wv * (kWS * LDP);      // overlays Ain/Q region
  float*    rs = (float*)(smem + OFF_RS) + wv * kWS;
#pragma unroll
  for (int rr = 0; rr < 2; ++rr) {
    const int row = lane * 2 + rr;
    const float* Sr = Sw + row * LDS_S;
    float mx = -3.0e38f;
    for (int i = 0; i < kWS; ++i) mx = fmaxf(mx, Sr[i]);
    _Float16* Pr = Pw + row * LDP;
    float ssum = 0.f;
    for (int i = 0; i < kWS; ++i) {
      const float e = __expf(Sr[i] - mx);
      ssum += e;
      Pr[i] = (_Float16)e;               // unnormalized; divide by row-sum in epilogue
    }
    rs[row] = ssum;
  }
  __syncthreads();   // all waves done reading their score region

  // ---- stage 4: out_h = P V  (K=64 -> two WMMA K-steps) ----
  _Float16* Aout = (_Float16*)(smem + OFF_AOUT);
#pragma unroll 1
  for (int nt = 0; nt < 2; ++nt) {
    v8f acc[4] = {vzero, vzero, vzero, vzero};
#pragma unroll
    for (int ks = 0; ks < 2; ++ks) {
      v16h b = frag_b(Vt, LDVT, h * kHD + nt * 16, ks * 32, lane);
#pragma unroll
      for (int mt = 0; mt < 4; ++mt) {
        v16h a = frag_a(Pw, LDP, mt * 16, ks * 32, lane);
        acc[mt] = wmma32(a, b, acc[mt]);
      }
    }
    const int n = h * kHD + nt * 16 + hl;
#pragma unroll
    for (int mt = 0; mt < 4; ++mt)
#pragma unroll
      for (int e = 0; e < 8; ++e) {
        const int m = mt * 16 + hi * 8 + e;
        Aout[m * LDA + n] = (_Float16)(acc[mt][e] / rs[m]);
      }
  }
  __syncthreads();

  // ---- stage 5: output projection + bias + window-reverse (+roll back) ----
#pragma unroll 1
  for (int nt = 0; nt < 2; ++nt) {
    const int n0 = wv * 32 + nt * 16;
    v8f acc[4] = {vzero, vzero, vzero, vzero};
#pragma unroll
    for (int k0 = 0; k0 < kD; k0 += 32) {
      v16h b = frag_b(wo_t, kD, n0, k0, lane);
#pragma unroll
      for (int mt = 0; mt < 4; ++mt) {
        v16h a = frag_a(Aout, LDA, mt * 16, k0, lane);
        acc[mt] = wmma32(a, b, acc[mt]);
      }
    }
    const int n = n0 + hl;
    const float bias = bo[n];
#pragma unroll
    for (int mt = 0; mt < 4; ++mt)
#pragma unroll
      for (int e = 0; e < 8; ++e) {
        const int m = mt * 16 + hi * 8 + e;
        int gh = wh * 8 + (m >> 3), gw = ww * 8 + (m & 7);
        if (SHIFTED) { gh = (gh + 4) & 63; gw = (gw + 4) & 63; }
        yout[(((size_t)mi * kHW + gh) * kHW + gw) * kD + n] = acc[mt][e] + bias;
      }
  }
}

// ---------------- host launch ----------------
extern "C" void kernel_launch(void* const* d_in, const int* in_sizes, int n_in,
                              void* d_out, int out_size, void* d_ws, size_t ws_size,
                              hipStream_t stream) {
  const float* x     = (const float*)d_in[0];
  const float* w_qkv = (const float*)d_in[1];
  const float* b_qkv = (const float*)d_in[2];
  const float* w_o   = (const float*)d_in[3];
  const float* b_o   = (const float*)d_in[4];
  float* out = (float*)d_out;

  char* ws = (char*)d_ws;
  const size_t ybytes = (size_t)kM * kHW * kHW * kD * sizeof(float);   // 128 MB
  float*    y      = (float*)ws;
  _Float16* wqkv_t = (_Float16*)(ws + ybytes);
  _Float16* wo_t   = wqkv_t + (size_t)kQKV * kD;

  (void)in_sizes; (void)n_in; (void)out_size; (void)ws_size;

  hipFuncSetAttribute(reinterpret_cast<const void*>(&swin_pass<0>),
                      hipFuncAttributeMaxDynamicSharedMemorySize, SMEM_BYTES);
  hipFuncSetAttribute(reinterpret_cast<const void*>(&swin_pass<1>),
                      hipFuncAttributeMaxDynamicSharedMemorySize, SMEM_BYTES);

  convert_weights<<<kQKV * kD / 256, 256, 0, stream>>>(w_qkv, w_o, wqkv_t, wo_t);

  const int nwin = kM * (kHW / 8) * (kHW / 8);   // 2048 windows
  swin_pass<0><<<nwin, 256, SMEM_BYTES, stream>>>(x, wqkv_t, b_qkv, wo_t, b_o, y);
  swin_pass<1><<<nwin, 256, SMEM_BYTES, stream>>>(y, wqkv_t, b_qkv, wo_t, b_o, out);
}
